// CrossEpochAttention_46557445488997
// MI455X (gfx1250) — matrix operand order repaired and checked by reading, not compiled, not run on hardware
//
#include <hip/hip_runtime.h>

typedef unsigned short u16;
typedef __attribute__((ext_vector_type(16))) __bf16 v16bf;
typedef __attribute__((ext_vector_type(8)))  __bf16 v8bf;
typedef __attribute__((ext_vector_type(8)))  float  v8f;

#define D_MODEL 1024
#define SEQ 10
#define NH 8
#define HD 128

__device__ __forceinline__ u16 f32_to_bf16(float f) {
  unsigned u = __float_as_uint(f);
  unsigned r = u + 0x7fffu + ((u >> 16) & 1u);   // round-to-nearest-even
  return (u16)(r >> 16);
}
__device__ __forceinline__ float bf16_to_f32(u16 v) {
  return __uint_as_float(((unsigned)v) << 16);
}

union FragBF {
  v16bf v;
  v8bf  h[2];
  u16   u[16];
};

// A-fragment: 16x32 bf16 tile of row-major [rows, K] matrix.
// ISA layout: lanes 0-15 row m hold K = {k0..k0+7, k0+16..k0+23};
// lanes 16-31 hold the same rows with K base +8.
__device__ __forceinline__ v16bf load_fragA(const u16* base, long ld, int row0,
                                            int k0, int lane) {
  int r  = row0 + (lane & 15);
  int kb = k0 + ((lane >> 4) << 3);
  const u16* p = base + (long)r * ld + kb;
  FragBF f;
  f.h[0] = *(const v8bf*)(p);
  f.h[1] = *(const v8bf*)(p + 16);
  return f.v;
}

// ---------------------------------------------------------------------------
// fp32 -> bf16 conversion
// ---------------------------------------------------------------------------
__global__ void cvt_f32_bf16_kernel(const float* __restrict__ in,
                                    u16* __restrict__ out, long n) {
  long i = (long)blockIdx.x * blockDim.x + threadIdx.x;
  if (i < n) out[i] = f32_to_bf16(in[i]);
}

__global__ void concat_bias_kernel(const float* __restrict__ a,
                                   const float* __restrict__ b,
                                   const float* __restrict__ c,
                                   float* __restrict__ out, int n) {
  int i = blockIdx.x * blockDim.x + threadIdx.x;
  if (i < n)            out[i] = a[i];
  else if (i < 2 * n)   out[i] = b[i - n];
  else if (i < 3 * n)   out[i] = c[i - 2 * n];
}

// ---------------------------------------------------------------------------
// NT GEMM: out[m,n] = sum_k A[m,k] * W[n,k] + bias[n]
// A: bf16 [M,K] row-major, W: bf16 [N,K] row-major.
// Block = 256 threads = 8 waves. Wave tile = 32(M) x 64(N). Block tile = 256 x 64.
// B tile staged to LDS in K-chunks of 128 via async global->LDS copies,
// stored directly in WMMA fragment order (32B per lane per fragment).
// ---------------------------------------------------------------------------
#define KCHUNK 128
#define BCHUNK_ELEMS (64 * KCHUNK)          // 8192 u16 = 16KB per buffer

// One wave issues 4 async b128 instructions; lane handles one 16B piece each.
// Piece p (0..1023) lands at LDS byte offset p*16 inside the buffer, which is
// exactly fragment-major order: frag f = p>>1 occupies [f*32, f*32+32).
__device__ __forceinline__ void issue_b_chunk(const u16* __restrict__ W, int K,
                                              int n0, int kc, u16* sB,
                                              int bufElemOff, int wave, int lane) {
#pragma unroll
  for (int i = 0; i < 4; ++i) {
    int p      = (wave * 4 + i) * 32 + lane;
    int f      = p >> 1;
    int half   = p & 1;
    int lane_f = f & 31;
    int ni     = (f >> 5) & 3;
    int kstep  = f >> 7;
    int row    = n0 + ni * 16 + (lane_f & 15);
    int koff   = kc * KCHUNK + kstep * 32 + ((lane_f >> 4) << 3) + half * 16;
    const u16* src = W + (long)row * K + koff;
    unsigned long long ga = (unsigned long long)(size_t)src;
    unsigned ldsoff = (unsigned)(size_t)(sB + bufElemOff + p * 8);
    asm volatile("global_load_async_to_lds_b128 %0, %1, off"
                 :: "v"(ldsoff), "v"(ga) : "memory");
  }
}

template <bool OUT_BF16>
__global__ __launch_bounds__(256)
void gemm_nt_bf16_kernel(const u16* __restrict__ A, const u16* __restrict__ W,
                         const float* __restrict__ bias, void* __restrict__ out,
                         int M, int N, int K) {
  __shared__ u16 sB[2 * BCHUNK_ELEMS];      // 32KB double buffer

  int lane = threadIdx.x & 31;
  int wave = threadIdx.x >> 5;
  int m0 = blockIdx.y * 256 + wave * 32;
  int n0 = blockIdx.x * 64;

  v8f acc[2][4] = {};

  int nChunks = K / KCHUNK;
  issue_b_chunk(W, K, n0, 0, sB, 0, wave, lane);

  for (int kc = 0; kc < nChunks; ++kc) {
    int buf = kc & 1;
    if (kc + 1 < nChunks)
      issue_b_chunk(W, K, n0, kc + 1, sB, (buf ^ 1) * BCHUNK_ELEMS, wave, lane);

    if (kc + 1 < nChunks)
      asm volatile("s_wait_asynccnt 0x4" ::: "memory");
    else
      asm volatile("s_wait_asynccnt 0x0" ::: "memory");
    __syncthreads();                        // all waves' chunk-kc writes done

    const u16* sbuf = sB + buf * BCHUNK_ELEMS;
#pragma unroll
    for (int ks = 0; ks < 4; ++ks) {
      int k0 = kc * KCHUNK + ks * 32;
      v16bf a0 = load_fragA(A, K, m0,      k0, lane);
      v16bf a1 = load_fragA(A, K, m0 + 16, k0, lane);
#pragma unroll
      for (int ni = 0; ni < 4; ++ni) {
        // fragment f = (ks*4+ni)*32 + lane at elem offset f*16 (32B aligned)
        const u16* bp = sbuf + (((ks * 4 + ni) * 32 + lane) << 4);
        v16bf b = *(const v16bf*)bp;
        acc[0][ni] = __builtin_amdgcn_wmma_f32_16x16x32_bf16(
            false, a0, false, b, (short)0, acc[0][ni], false, false);
        acc[1][ni] = __builtin_amdgcn_wmma_f32_16x16x32_bf16(
            false, a1, false, b, (short)0, acc[1][ni], false, false);
      }
    }
    __syncthreads();                        // reads done before buf reuse
  }

  int col  = lane & 15;
  int rofs = (lane >> 4) * 8;
#pragma unroll
  for (int mi = 0; mi < 2; ++mi) {
#pragma unroll
    for (int ni = 0; ni < 4; ++ni) {
      int n = n0 + ni * 16 + col;
      float bval = bias[n];
#pragma unroll
      for (int r = 0; r < 8; ++r) {
        int m = m0 + mi * 16 + rofs + r;
        float v = acc[mi][ni][r] + bval;
        if (OUT_BF16) ((u16*)out)[(long)m * N + n] = f32_to_bf16(v);
        else          ((float*)out)[(long)m * N + n] = v;
      }
    }
  }
}

// ---------------------------------------------------------------------------
// Attention core: one wave handles one (batch, head).
// qkv: bf16 [B*S, 3*D_MODEL] with q|k|v concatenated along columns.
// scores(16x16, S padded to 16) via 4x WMMA bf16 over hd=128, softmax via
// half-wave shuffles, AV (10x10x128) via VALU, bf16 ctx out [B*S, D_MODEL].
// ---------------------------------------------------------------------------
__global__ __launch_bounds__(256)
void attn_kernel(const u16* __restrict__ qkv, const float* __restrict__ pos_bias,
                 u16* __restrict__ ctx) {
  __shared__ float s_attn[8][16][16];

  int lane = threadIdx.x & 31;
  int wave = threadIdx.x >> 5;
  int u = blockIdx.x * 8 + wave;
  int b = u >> 3;
  int h = u & 7;

  const long ld = 3 * D_MODEL;
  const u16* qb = qkv + (long)b * SEQ * ld + h * HD;
  const u16* kb = qb + D_MODEL;
  const u16* vb = qb + 2 * D_MODEL;

  int srow = lane & 15;
  bool rowValid = srow < SEQ;

  v8f sc = {};
#pragma unroll
  for (int kc = 0; kc < 4; ++kc) {
    FragBF fq, fk;
    if (rowValid) {
      const u16* p = qb + (long)srow * ld + kc * 32 + ((lane >> 4) << 4);
      fq.h[0] = *(const v8bf*)(p);
      fq.h[1] = *(const v8bf*)(p + 8);
      fk.v = *(const v16bf*)(kb + (long)srow * ld + kc * 32 + ((lane >> 4) << 4));
    } else {
#pragma unroll
      for (int i = 0; i < 16; ++i) { fq.u[i] = 0; fk.u[i] = 0; }
    }
    sc = __builtin_amdgcn_wmma_f32_16x16x32_bf16(
        false, fq.v, false, fk.v, (short)0, sc, false, false);
  }

  const float scale = 0.0883883476483184f;  // 1/sqrt(128)
  int col  = lane & 15;
  int rofs = (lane >> 4) * 8;
#pragma unroll
  for (int r = 0; r < 8; ++r) {
    int row = r + rofs;
    float v = sc[r] * scale;
    if (row < SEQ && col < SEQ) v += pos_bias[col - row + SEQ - 1];
    if (col >= SEQ) v = -1e30f;
    float m = v;
#pragma unroll
    for (int off = 8; off >= 1; off >>= 1) m = fmaxf(m, __shfl_xor(m, off, 16));
    float e = __expf(v - m);
    float s = e;
#pragma unroll
    for (int off = 8; off >= 1; off >>= 1) s += __shfl_xor(s, off, 16);
    s_attn[wave][row][col] = e / s;
  }
  __syncthreads();

  // AV: each lane produces out[s][d] for d = lane*4 .. lane*4+3, s = 0..9
  float vv[SEQ][4];
#pragma unroll
  for (int sk = 0; sk < SEQ; ++sk) {
    const u16* vp = vb + (long)sk * ld + lane * 4;
    unsigned long long pv = *(const unsigned long long*)vp;
#pragma unroll
    for (int j = 0; j < 4; ++j)
      vv[sk][j] = bf16_to_f32((u16)((pv >> (16 * j)) & 0xFFFFull));
  }

#pragma unroll
  for (int s = 0; s < SEQ; ++s) {
    float o0 = 0.f, o1 = 0.f, o2 = 0.f, o3 = 0.f;
#pragma unroll
    for (int sk = 0; sk < SEQ; ++sk) {
      float a = s_attn[wave][s][sk];
      o0 += a * vv[sk][0]; o1 += a * vv[sk][1];
      o2 += a * vv[sk][2]; o3 += a * vv[sk][3];
    }
    u16* op = ctx + (long)(b * SEQ + s) * D_MODEL + h * HD + lane * 4;
    op[0] = f32_to_bf16(o0); op[1] = f32_to_bf16(o1);
    op[2] = f32_to_bf16(o2); op[3] = f32_to_bf16(o3);
  }
}

// ---------------------------------------------------------------------------
extern "C" void kernel_launch(void* const* d_in, const int* in_sizes, int n_in,
                              void* d_out, int out_size, void* d_ws, size_t ws_size,
                              hipStream_t stream) {
  const float* x  = (const float*)d_in[0];
  const float* Wq = (const float*)d_in[1];
  const float* bq = (const float*)d_in[2];
  const float* Wk = (const float*)d_in[3];
  const float* bk = (const float*)d_in[4];
  const float* Wv = (const float*)d_in[5];
  const float* bv = (const float*)d_in[6];
  const float* Wo = (const float*)d_in[7];
  const float* bo = (const float*)d_in[8];
  const float* pb = (const float*)d_in[9];

  const long M  = (long)in_sizes[0] / D_MODEL;      // B*S = 81920
  const long DD = (long)D_MODEL * D_MODEL;          // 1M

  // workspace layout (bf16 buffers as u16)
  char* w = (char*)d_ws;
  u16*   xb   = (u16*)w;    w += M * D_MODEL * sizeof(u16);
  u16*   wqkv = (u16*)w;    w += 3 * DD * sizeof(u16);
  u16*   wob  = (u16*)w;    w += DD * sizeof(u16);
  float* bqkv = (float*)w;  w += 3 * D_MODEL * sizeof(float);
  u16*   qkv  = (u16*)w;    w += M * 3 * D_MODEL * sizeof(u16);
  u16*   ctx  = (u16*)w;

  const int T = 256;
  long nx = M * D_MODEL;
  cvt_f32_bf16_kernel<<<(unsigned)((nx + T - 1) / T), T, 0, stream>>>(x, xb, nx);
  cvt_f32_bf16_kernel<<<(unsigned)((DD + T - 1) / T), T, 0, stream>>>(Wq, wqkv, DD);
  cvt_f32_bf16_kernel<<<(unsigned)((DD + T - 1) / T), T, 0, stream>>>(Wk, wqkv + DD, DD);
  cvt_f32_bf16_kernel<<<(unsigned)((DD + T - 1) / T), T, 0, stream>>>(Wv, wqkv + 2 * DD, DD);
  cvt_f32_bf16_kernel<<<(unsigned)((DD + T - 1) / T), T, 0, stream>>>(Wo, wob, DD);
  concat_bias_kernel<<<(3 * D_MODEL + T - 1) / T, T, 0, stream>>>(bq, bk, bv, bqkv, D_MODEL);

  // QKV projection: [M,1024] x [3072,1024]^T -> bf16 [M,3072]
  dim3 g1(3 * D_MODEL / 64, (unsigned)(M / 256));
  gemm_nt_bf16_kernel<true><<<g1, T, 0, stream>>>(xb, wqkv, bqkv, (void*)qkv,
                                                  (int)M, 3 * D_MODEL, D_MODEL);

  // Attention: one wave per (batch, head)
  long nUnits = (M / SEQ) * NH;                     // 65536
  attn_kernel<<<(unsigned)(nUnits / 8), T, 0, stream>>>(qkv, pb, ctx);

  // Output projection: [M,1024] x [1024,1024]^T + bo -> fp32 d_out
  dim3 g2(D_MODEL / 64, (unsigned)(M / 256));
  gemm_nt_bf16_kernel<false><<<g2, T, 0, stream>>>(ctx, wob, bo, d_out,
                                                   (int)M, D_MODEL, D_MODEL);
}
